// Encoder_42210938585608
// MI455X (gfx1250) — matrix operand order repaired and checked by reading, not compiled, not run on hardware
//
#include <hip/hip_runtime.h>

// ---------------------------------------------------------------------------
// 2-layer GCN forward for MI455X (gfx1250, wave32).
//  - GEMMs: v_wmma_f32_16x16x32_bf16, LDS tiles fed by global_load_async_to_lds_b128
//    (ASYNCcnt) with double-buffered LDS so copy of tile i+1 overlaps WMMA on tile i.
//  - Inputs pre-converted to bf16 (and B pre-transposed) so the hot loop has zero
//    conversion VALU work and zero bounds checks (row counts padded to 128).
//  - Scatter-add: f32 global atomics; agg working sets (102/51 MB) fit 192MB L2.
// ---------------------------------------------------------------------------

typedef __bf16 bf16_t;
typedef __attribute__((ext_vector_type(16))) __bf16 v16bf;
typedef __attribute__((ext_vector_type(8)))  float  v8f;

union Frag { v16bf v; uint4 q[2]; };

__device__ __forceinline__ unsigned short f2bfu(float f) {
    unsigned u = __float_as_uint(f);
    u += 0x7FFFu + ((u >> 16) & 1u);           // round-to-nearest-even
    return (unsigned short)(u >> 16);
}
__device__ __forceinline__ bf16_t f2bf(float f) {
    unsigned short h = f2bfu(f);
    bf16_t r;
    __builtin_memcpy(&r, &h, sizeof(r));
    return r;
}
__device__ __forceinline__ float bfu2f(unsigned s) {
    return __uint_as_float(s << 16);
}

// ---------------------------------------------------------------------------
// Tiled bf16 WMMA GEMM, async-LDS pipelined:
//   C[Mpad,N] (bf16) = A[Mpad,K] (bf16) * Bt[N,K] (bf16, pre-transposed)
// Block: 256 threads = 8 waves; block tile 128x128, BK=32, double-buffered LDS.
// Wave grid 4(rows)x2(cols): each wave owns 32x64 = 2x4 WMMA tiles.
// ---------------------------------------------------------------------------
#define BM 128
#define BN 128
#define BK 32
#define LDT 40   // padded LDS row stride in bf16 (80B, 16B-aligned)

__global__ __launch_bounds__(256)
void gemm_bf16_async(const bf16_t* __restrict__ A, const bf16_t* __restrict__ Bt,
                     bf16_t* __restrict__ C, int Mpad, int Nn, int K) {
    __shared__ __align__(16) bf16_t As[2][BM * LDT];   // [buf][row][k]
    __shared__ __align__(16) bf16_t Bs[2][BN * LDT];   // [buf][col][k]

    const int t     = threadIdx.x;
    const int lane  = t & 31;
    const int wave  = t >> 5;
    const int wr    = wave & 3;          // wave row: 32 rows each
    const int wc    = wave >> 2;         // wave col: 64 cols each
    const int row0  = blockIdx.y * BM;
    const int col0  = blockIdx.x * BN;
    const int lhalf = lane >> 4;
    const int l15   = lane & 15;

    // Per-thread async copy chunks: tile = 128 rows x 32 bf16 = 512 x 16B chunks.
    // Thread t owns chunks t (row r0) and t+256 (row r0+64), same 8-bf16 column c0.
    const int r0 = t >> 2;
    const int c0 = (t & 3) << 3;
    const bf16_t* gA = A  + (size_t)(row0 + r0) * K + c0;
    const bf16_t* gB = Bt + (size_t)(col0 + r0) * K + c0;
    const unsigned ldsChunk = (unsigned)((r0 * LDT + c0) * 2);      // byte offset in tile
    const unsigned ldsChunk2 = ldsChunk + (unsigned)(64 * LDT * 2); // row r0+64
    unsigned asB[2], bsB[2];
    asB[0] = (unsigned)(size_t)&As[0][0];  asB[1] = (unsigned)(size_t)&As[1][0];
    bsB[0] = (unsigned)(size_t)&Bs[0][0];  bsB[1] = (unsigned)(size_t)&Bs[1][0];

    auto issue = [&](int k0, int buf) {
        const size_t rstep = (size_t)64 * K;
        const void* pa0 = (const void*)(gA + k0);
        const void* pa1 = (const void*)(gA + rstep + k0);
        const void* pb0 = (const void*)(gB + k0);
        const void* pb1 = (const void*)(gB + rstep + k0);
        unsigned a0 = asB[buf] + ldsChunk,  a1 = asB[buf] + ldsChunk2;
        unsigned b0 = bsB[buf] + ldsChunk,  b1 = bsB[buf] + ldsChunk2;
        asm volatile("global_load_async_to_lds_b128 %0, %1, off" :: "v"(a0), "v"(pa0) : "memory");
        asm volatile("global_load_async_to_lds_b128 %0, %1, off" :: "v"(a1), "v"(pa1) : "memory");
        asm volatile("global_load_async_to_lds_b128 %0, %1, off" :: "v"(b0), "v"(pb0) : "memory");
        asm volatile("global_load_async_to_lds_b128 %0, %1, off" :: "v"(b1), "v"(pb1) : "memory");
    };

    v8f acc[2][4];
    const v8f vzero = {0.f, 0.f, 0.f, 0.f, 0.f, 0.f, 0.f, 0.f};
#pragma unroll
    for (int mi = 0; mi < 2; ++mi)
#pragma unroll
        for (int ni = 0; ni < 4; ++ni) acc[mi][ni] = vzero;

    const int nk = K / BK;
    issue(0, 0);

    for (int i = 0; i < nk; ++i) {
        const int cur = i & 1;
        asm volatile("s_wait_asynccnt 0" ::: "memory");   // own tile-i chunks landed
        __syncthreads();                                  // publish tile i; prev reads done
        if (i + 1 < nk) issue((i + 1) * BK, 1 - cur);     // overlap copy with WMMA below

        // Fragment loads (two ds_load_b128 each), per the 16-bit A/B VGPR layouts.
        Frag a[2], b[4];
        const bf16_t* Ab = As[cur];
        const bf16_t* Bb = Bs[cur];
        const int klo = lhalf * 8;    // A: lanes 0-15 -> K0..7/16..23; 16-31 -> K8..15/24..31
        const int kb  = lhalf * 8;    // B: lanes 0-15 -> K0..7/16..23; 16-31 -> K8..15/24..31
#pragma unroll
        for (int mi = 0; mi < 2; ++mi) {
            const bf16_t* p = Ab + (wr * 32 + mi * 16 + l15) * LDT;
            a[mi].q[0] = *(const uint4*)(p + klo);
            a[mi].q[1] = *(const uint4*)(p + klo + 16);
        }
#pragma unroll
        for (int ni = 0; ni < 4; ++ni) {
            const bf16_t* p = Bb + (wc * 64 + ni * 16 + l15) * LDT;
            b[ni].q[0] = *(const uint4*)(p + kb);
            b[ni].q[1] = *(const uint4*)(p + kb + 16);
        }
#pragma unroll
        for (int mi = 0; mi < 2; ++mi)
#pragma unroll
            for (int ni = 0; ni < 4; ++ni)
                acc[mi][ni] = __builtin_amdgcn_wmma_f32_16x16x32_bf16(
                    false, a[mi].v, false, b[ni].v, (short)0, acc[mi][ni],
                    false, false);
    }

    // Store C (bf16). D layout: VGPR v -> row v + 8*lhalf, col = l15 per 16x16 tile.
#pragma unroll
    for (int mi = 0; mi < 2; ++mi) {
        int rbase = row0 + wr * 32 + mi * 16 + lhalf * 8;
#pragma unroll
        for (int ni = 0; ni < 4; ++ni) {
            int col = col0 + wc * 64 + ni * 16 + l15;
#pragma unroll
            for (int v = 0; v < 8; ++v)
                C[(size_t)(rbase + v) * Nn + col] = f2bf(acc[mi][ni][v]);
        }
    }
}

// ---------------------------------------------------------------------------
// Pre/post-pass kernels
// ---------------------------------------------------------------------------
__global__ void fill_f32(float* __restrict__ p, float v, long long n) {
    long long i = (long long)blockIdx.x * blockDim.x + threadIdx.x;
    if (i < n) p[i] = v;
}

// dst[r][0..F) = bf16(src[r][..]) for r<rows, 0 for rows<=r<rowsPad. clog2 = log2(F/4)
__global__ void cvt_pad_bf16(const float* __restrict__ src, bf16_t* __restrict__ dst,
                             int rows, int rowsPad, int F, int clog2) {
    long long i = (long long)blockIdx.x * blockDim.x + threadIdx.x;
    if (i >= ((long long)rowsPad << clog2)) return;
    const int row = (int)(i >> clog2);
    const int c   = (int)(i & ((1 << clog2) - 1)) << 2;
    float4 v = make_float4(0.f, 0.f, 0.f, 0.f);
    if (row < rows) v = *(const float4*)(src + (size_t)row * F + c);
    uint2 o;
    o.x = (unsigned)f2bfu(v.x) | ((unsigned)f2bfu(v.y) << 16);
    o.y = (unsigned)f2bfu(v.z) | ((unsigned)f2bfu(v.w) << 16);
    *(uint2*)((void*)(dst + (size_t)row * F + c)) = o;
}

// Wt[n][k] = bf16(W[k][n])
__global__ void transpose_to_bf16(const float* __restrict__ W, bf16_t* __restrict__ Wt,
                                  int K, int Nn) {
    int i = blockIdx.x * blockDim.x + threadIdx.x;
    if (i >= K * Nn) return;
    int n = i / K, k = i - n * K;
    Wt[(size_t)n * K + k] = f2bf(W[(size_t)k * Nn + n]);
}

__global__ void deg_accum(const long long* __restrict__ ei, float* __restrict__ deg, int E) {
    int e = blockIdx.x * blockDim.x + threadIdx.x;
    if (e < E) atomicAdd(deg + (int)ei[E + e], 1.0f);
}

__global__ void rsqrt_inplace(float* __restrict__ p, int n) {
    int i = blockIdx.x * blockDim.x + threadIdx.x;
    if (i < n) p[i] = rsqrtf(p[i]);
}

// agg[dst] += bf16(h[src]) * dinv[src]*dinv[dst]; (1<<clog2) float4-chunks per edge
__global__ void scatter_add(const bf16_t* __restrict__ h, const long long* __restrict__ ei,
                            const float* __restrict__ dinv, float* __restrict__ agg,
                            int E, int F, int clog2) {
    const int chunks = 1 << clog2;
    const int epb    = blockDim.x >> clog2;
    const int e = blockIdx.x * epb + (threadIdx.x >> clog2);
    if (e >= E) return;
    const int c = (threadIdx.x & (chunks - 1)) << 2;
    const int s = (int)ei[e];
    const int d = (int)ei[E + e];
    const float nrm = dinv[s] * dinv[d];
    const uint2 raw = *(const uint2*)((const void*)(h + (size_t)s * F + c));
    float* base = agg + (size_t)d * F + c;
    atomicAdd(base + 0, bfu2f(raw.x & 0xffffu) * nrm);
    atomicAdd(base + 1, bfu2f(raw.x >> 16)     * nrm);
    atomicAdd(base + 2, bfu2f(raw.y & 0xffffu) * nrm);
    atomicAdd(base + 3, bfu2f(raw.y >> 16)     * nrm);
}

// out = relu(agg + h*dinv^2 + bias); h is bf16; OUTBF selects bf16 vs f32 output
template <bool OUTBF>
__global__ void combine_relu(const float* __restrict__ agg, const bf16_t* __restrict__ h,
                             const float* __restrict__ dinv, const float* __restrict__ bias,
                             void* __restrict__ out, int Nn, int F, int clog2) {
    long long i = (long long)blockIdx.x * blockDim.x + threadIdx.x;
    if (i >= ((long long)Nn << clog2)) return;
    const int row = (int)(i >> clog2);
    const int c   = (int)(i & ((1 << clog2) - 1)) << 2;
    const float di = dinv[row];
    const float dd = di * di;
    const float4 a  = *(const float4*)(agg + (size_t)row * F + c);
    const uint2 raw = *(const uint2*)((const void*)(h + (size_t)row * F + c));
    const float4 b  = *(const float4*)(bias + c);
    float o0 = fmaxf(a.x + bfu2f(raw.x & 0xffffu) * dd + b.x, 0.f);
    float o1 = fmaxf(a.y + bfu2f(raw.x >> 16)     * dd + b.y, 0.f);
    float o2 = fmaxf(a.z + bfu2f(raw.y & 0xffffu) * dd + b.z, 0.f);
    float o3 = fmaxf(a.w + bfu2f(raw.y >> 16)     * dd + b.w, 0.f);
    if (OUTBF) {
        uint2 o;
        o.x = (unsigned)f2bfu(o0) | ((unsigned)f2bfu(o1) << 16);
        o.y = (unsigned)f2bfu(o2) | ((unsigned)f2bfu(o3) << 16);
        *(uint2*)((void*)((bf16_t*)out + (size_t)row * F + c)) = o;
    } else {
        float4 o = make_float4(o0, o1, o2, o3);
        *(float4*)((float*)out + (size_t)row * F + c) = o;
    }
}

// ---------------------------------------------------------------------------
// Launch
// ---------------------------------------------------------------------------
extern "C" void kernel_launch(void* const* d_in, const int* in_sizes, int n_in,
                              void* d_out, int out_size, void* d_ws, size_t ws_size,
                              hipStream_t stream) {
    const float*     x  = (const float*)d_in[0];
    const long long* ei = (const long long*)d_in[1];   // int64 edge_index [2][E]
    const float*     W1 = (const float*)d_in[2];
    const float*     b1 = (const float*)d_in[3];
    const float*     W2 = (const float*)d_in[4];
    const float*     b2 = (const float*)d_in[5];

    const int IN = 128, HID = 256, OUT = 128;
    const int N    = in_sizes[0] / IN;                 // 100000
    const int E    = in_sizes[1] / 2;                  // 1600000
    const int Npad = ((N + BM - 1) / BM) * BM;         // 100096

    // workspace layout (all regions 256B-aligned)
    float*  ws     = (float*)d_ws;
    float*  dinv   = ws;                                          // Npad f32
    bf16_t* xbf    = (bf16_t*)(ws + Npad);                        // Npad*IN bf16
    bf16_t* w1t    = xbf + (size_t)Npad * IN;                     // HID*IN  bf16 (transposed)
    bf16_t* w2t    = w1t + (size_t)HID * IN;                      // OUT*HID bf16 (transposed)
    bf16_t* hlinbf = w2t + (size_t)OUT * HID;                     // Npad*HID bf16 (h_lin)
    bf16_t* h1bf   = hlinbf + (size_t)Npad * HID;                 // Npad*HID bf16 (relu h1)
    float*  aggbuf = (float*)(h1bf + (size_t)Npad * HID);         // Npad*HID f32
    float*  agg2   = aggbuf;                                      // Npad*OUT f32 view
    bf16_t* h2lbf  = (bf16_t*)(aggbuf + (size_t)Npad * OUT);      // Npad*OUT bf16 view
    float*  out    = (float*)d_out;

    // ---- pre-pass: bf16 copies (padded) + transposed weights ----
    {
        long long n = (long long)Npad * (IN >> 2);
        cvt_pad_bf16<<<(int)((n + 255) / 256), 256, 0, stream>>>(x, xbf, N, Npad, IN, 5);
        transpose_to_bf16<<<(IN * HID + 255) / 256, 256, 0, stream>>>(W1, w1t, IN, HID);
        transpose_to_bf16<<<(HID * OUT + 255) / 256, 256, 0, stream>>>(W2, w2t, HID, OUT);
    }

    // ---- degree (with self loop) -> dinv = rsqrt(deg) ----
    fill_f32<<<(N + 255) / 256, 256, 0, stream>>>(dinv, 1.0f, N);
    deg_accum<<<(E + 255) / 256, 256, 0, stream>>>(ei, dinv, E);
    rsqrt_inplace<<<(N + 255) / 256, 256, 0, stream>>>(dinv, N);

    // ---- layer 1 ----
    {
        long long nz = (long long)N * HID;
        fill_f32<<<(int)((nz + 255) / 256), 256, 0, stream>>>(aggbuf, 0.0f, nz);
    }
    gemm_bf16_async<<<dim3(HID / BN, Npad / BM), 256, 0, stream>>>(
        xbf, w1t, hlinbf, Npad, HID, IN);
    scatter_add<<<(E + 3) / 4, 256, 0, stream>>>(hlinbf, ei, dinv, aggbuf, E, HID, 6);
    {   // zero pad rows of h1bf (layer-2 GEMM reads Npad rows)
        long long nf = (long long)(Npad - N) * (HID / 2);
        if (nf > 0)
            fill_f32<<<(int)((nf + 255) / 256), 256, 0, stream>>>(
                (float*)(h1bf + (size_t)N * HID), 0.0f, nf);
    }
    combine_relu<true><<<(int)((((long long)N << 6) + 255) / 256), 256, 0, stream>>>(
        aggbuf, hlinbf, dinv, b1, (void*)h1bf, N, HID, 6);

    // ---- layer 2 ----
    {
        long long nz = (long long)N * OUT;
        fill_f32<<<(int)((nz + 255) / 256), 256, 0, stream>>>(agg2, 0.0f, nz);
    }
    gemm_bf16_async<<<dim3(OUT / BN, Npad / BM), 256, 0, stream>>>(
        h1bf, w2t, h2lbf, Npad, OUT, HID);
    scatter_add<<<(E + 7) / 8, 256, 0, stream>>>(h2lbf, ei, dinv, agg2, E, OUT, 5);
    combine_relu<false><<<(int)((((long long)N << 5) + 255) / 256), 256, 0, stream>>>(
        agg2, h2lbf, dinv, b2, (void*)out, N, OUT, 5);
}
